// EncoderBlock_24618752540941
// MI455X (gfx1250) — matrix operand order repaired and checked by reading, not compiled, not run on hardware
//
#include <hip/hip_runtime.h>
#include <math.h>
#include <stdint.h>

// ---------------------------------------------------------------------------
// Problem constants (from the reference)
// ---------------------------------------------------------------------------
#define DMODEL 1024
#define NHEADS 16
#define DK     64
#define NF     7          // ceil(ln(1024)) = 7
#define DFF    4096
#define BB     2
#define SS     4096
#define ROWS   (BB*SS)    // 8192

typedef __attribute__((ext_vector_type(16))) __bf16 v16bf;
typedef __attribute__((ext_vector_type(8)))  float  v8f;

// ---------------------------------------------------------------------------
// CDNA5 async global->LDS copy (ASYNCcnt-tracked DMA, no VGPR round trip).
// Signature (from hipcc diagnostic): takes v4i pointers, global src in AS(1),
// LDS dst in AS(3), plus imm offset and imm cpol.
// Falls back to a synchronous float4 copy if the toolchain lacks the builtin.
// ---------------------------------------------------------------------------
#if defined(__has_builtin)
#if __has_builtin(__builtin_amdgcn_global_load_async_to_lds_b128)
#define USE_ASYNC_LDS 1
#endif
#endif
#ifndef USE_ASYNC_LDS
#define USE_ASYNC_LDS 0
#endif

typedef int v4i_vs __attribute__((vector_size(4 * sizeof(int))));
typedef __attribute__((address_space(1))) v4i_vs* g_v4i_p;
typedef __attribute__((address_space(3))) v4i_vs* l_v4i_p;

__device__ inline void copy16_g2l(const __bf16* g, __bf16* l) {
#if USE_ASYNC_LDS
  __builtin_amdgcn_global_load_async_to_lds_b128(
      (g_v4i_p)(uintptr_t)g, (l_v4i_p)(uintptr_t)l, 0, 0);
#else
  *(float4*)l = *(const float4*)g;
#endif
}

__device__ inline void wait_async_copies() {
#if USE_ASYNC_LDS
#if defined(__has_builtin) && __has_builtin(__builtin_amdgcn_s_wait_asynccnt)
  __builtin_amdgcn_s_wait_asynccnt(0);
#else
  asm volatile("s_wait_asynccnt 0x0" ::: "memory");
#endif
#endif
}

// ---------------------------------------------------------------------------
// f32 -> bf16 conversion (weights, once per launch)
// ---------------------------------------------------------------------------
__global__ __launch_bounds__(256)
void f2b_kernel(const float* __restrict__ in, __bf16* __restrict__ out, int n) {
  int i = blockIdx.x * 256 + threadIdx.x;
  if (i < n) out[i] = (__bf16)in[i];
}

// ---------------------------------------------------------------------------
// LayerNorm (ddof=1 std, eps added to std) -> bf16 output for WMMA GEMMs
// ---------------------------------------------------------------------------
__global__ __launch_bounds__(256)
void layernorm_kernel(const float* __restrict__ x,
                      const float* __restrict__ alpha,
                      const float* __restrict__ beta,
                      __bf16* __restrict__ out) {
  __shared__ float red[256];
  const int row = blockIdx.x;
  const float* xr = x + (size_t)row * DMODEL;
  const int t = threadIdx.x;

  float s = 0.f;
  for (int i = t; i < DMODEL; i += 256) s += xr[i];
  red[t] = s; __syncthreads();
  for (int off = 128; off > 0; off >>= 1) { if (t < off) red[t] += red[t + off]; __syncthreads(); }
  const float mean = red[0] * (1.0f / DMODEL);
  __syncthreads();

  float v = 0.f;
  for (int i = t; i < DMODEL; i += 256) { float d = xr[i] - mean; v += d * d; }
  red[t] = v; __syncthreads();
  for (int off = 128; off > 0; off >>= 1) { if (t < off) red[t] += red[t + off]; __syncthreads(); }
  const float stdv = sqrtf(red[0] * (1.0f / (DMODEL - 1)));
  const float inv = 1.0f / (stdv + 1e-6f);

  __bf16* orow = out + (size_t)row * DMODEL;
  for (int i = t; i < DMODEL; i += 256)
    orow[i] = (__bf16)(alpha[i] * (xr[i] - mean) * inv + beta[i]);
}

// ---------------------------------------------------------------------------
// bf16 WMMA GEMM:  C[M,N] = A[M,K] @ W[N,K]^T  (+bias, gelu, residual)
// Tile 128x64x32, 256 threads = 8 waves, double-buffered LDS filled by
// GLOBAL_LOAD_ASYNC_TO_LDS_B128. Each wave owns a 2x2 grid of 16x16 tiles:
// 4 WMMAs per K-step with A/B fragments each reused twice.
// LDS rows padded to 40 bf16 (80B) so every fragment read is an aligned
// ds_load_b128 matching the ISA VGPR layouts:
//   A 16x32 bf16 : lane l -> row l&15, chunks at k = 8*(l>>4) and 16+8*(l>>4)
//   B 32x16 bf16 : lane l -> col l&15, chunks at k = 8*(l>>4) and 16+8*(l>>4)
//   C 16x16 f32  : lane l -> col l&15, row = 8*(l>>4) + r
// ---------------------------------------------------------------------------
__device__ inline float gelu_f(float x) {
  const float c = 0.7978845608028654f;  // sqrt(2/pi)
  return 0.5f * x * (1.0f + tanhf(c * (x + 0.044715f * x * x * x)));
}

template<bool HAS_BIAS, bool DO_GELU, bool HAS_RES, bool OUT_F32, bool OUT_BF16>
__global__ __launch_bounds__(256)
void gemm_bf16_kernel(const __bf16* __restrict__ A,
                      const __bf16* __restrict__ W,
                      const float* __restrict__ bias,
                      const float* __restrict__ res,
                      float* __restrict__ outf,
                      __bf16* __restrict__ outb,
                      int M, int N, int K) {
  constexpr int BM = 128, BN = 64, BK = 32, LDT = 40; // LDT*2 = 80B (16B mult.)
  __shared__ __bf16 sA[2][BM * LDT];
  __shared__ __bf16 sW[2][BN * LDT];

  const int tid  = threadIdx.x;
  const int lane = tid & 31;
  const int wave = tid >> 5;
  const int half = lane >> 4;
  const int l16  = lane & 15;
  const int bm = blockIdx.x * BM;
  const int bn = blockIdx.y * BN;
  const int wm0 = (wave >> 1) * 32;   // wave's 2 M-tiles: wm0, wm0+16
  const int wn0 = (wave & 1) * 32;    // wave's 2 N-tiles: wn0, wn0+16

  const int ldr = tid >> 2;           // 0..63
  const int ldc = (tid & 3) * 8;      // 0,8,16,24 (8 bf16 = 16B)

  v8f acc00 = {}, acc01 = {}, acc10 = {}, acc11 = {};

  auto issue_tile = [&](int buf, int k0) {
    // A tile: 128x32 -> two 16B chunks per thread; W tile: 64x32 -> one chunk
    copy16_g2l(A + (size_t)(bm + ldr) * K + k0 + ldc,      &sA[buf][ldr * LDT + ldc]);
    copy16_g2l(A + (size_t)(bm + 64 + ldr) * K + k0 + ldc, &sA[buf][(64 + ldr) * LDT + ldc]);
    copy16_g2l(W + (size_t)(bn + ldr) * K + k0 + ldc,      &sW[buf][ldr * LDT + ldc]);
  };

  const int NT = K / BK;
  issue_tile(0, 0);

  for (int kt = 0; kt < NT; ++kt) {
    const int cur = kt & 1;
    wait_async_copies();   // this wave's async fills of buf `cur` have landed
    __syncthreads();       // whole block's fills done + prior reads of cur^1 done
    if (kt + 1 < NT) issue_tile(cur ^ 1, (kt + 1) * BK);

    union Frag { v16bf v; float4 f4[2]; };
    Frag fa0, fa1, fb0, fb1;
    {
      const __bf16* base = &sA[cur][0];
      const int r0 = (wm0 + l16) * LDT, r1 = (wm0 + 16 + l16) * LDT;
      fa0.f4[0] = *(const float4*)(base + r0 + half * 8);
      fa0.f4[1] = *(const float4*)(base + r0 + 16 + half * 8);
      fa1.f4[0] = *(const float4*)(base + r1 + half * 8);
      fa1.f4[1] = *(const float4*)(base + r1 + 16 + half * 8);
    }
    {
      const __bf16* base = &sW[cur][0];
      const int r0 = (wn0 + l16) * LDT, r1 = (wn0 + 16 + l16) * LDT;
      fb0.f4[0] = *(const float4*)(base + r0 + half * 8);
      fb0.f4[1] = *(const float4*)(base + r0 + 16 + half * 8);
      fb1.f4[0] = *(const float4*)(base + r1 + half * 8);
      fb1.f4[1] = *(const float4*)(base + r1 + 16 + half * 8);
    }

    acc00 = __builtin_amdgcn_wmma_f32_16x16x32_bf16(false, fa0.v, false, fb0.v, (short)0, acc00, false, false);
    acc01 = __builtin_amdgcn_wmma_f32_16x16x32_bf16(false, fa0.v, false, fb1.v, (short)0, acc01, false, false);
    acc10 = __builtin_amdgcn_wmma_f32_16x16x32_bf16(false, fa1.v, false, fb0.v, (short)0, acc10, false, false);
    acc11 = __builtin_amdgcn_wmma_f32_16x16x32_bf16(false, fa1.v, false, fb1.v, (short)0, acc11, false, false);
    __syncthreads();       // reads of buf `cur` done before it is refilled
  }

  // Epilogue: D layout -> lane l16 = column, VGPR r -> row 8*half + r
  v8f accs[2][2] = { { acc00, acc01 }, { acc10, acc11 } };
  #pragma unroll
  for (int mi = 0; mi < 2; ++mi) {
    #pragma unroll
    for (int ni = 0; ni < 2; ++ni) {
      const v8f acc = accs[mi][ni];
      const int n = bn + wn0 + ni * 16 + l16;
      float bval = 0.f;
      if constexpr (HAS_BIAS) bval = bias[n];
      #pragma unroll
      for (int r = 0; r < 8; ++r) {
        const int m = bm + wm0 + mi * 16 + half * 8 + r;
        float val = acc[r] + bval;
        if constexpr (DO_GELU) val = gelu_f(val);
        const size_t idx = (size_t)m * N + n;
        if constexpr (HAS_RES)  val += res[idx];
        if constexpr (OUT_F32)  outf[idx] = val;
        if constexpr (OUT_BF16) outb[idx] = (__bf16)val;
      }
    }
  }
}

// ---------------------------------------------------------------------------
// Performer feature map: qp = exp(-0.5*(q @ omega.T)^2), normalized over NF
// ---------------------------------------------------------------------------
__global__ __launch_bounds__(256)
void phi_kernel(const float* __restrict__ q,
                const float* __restrict__ omega,   // [NF, DK]
                float* __restrict__ qp) {          // [ROWS, NHEADS, NF]
  int idx = blockIdx.x * 256 + threadIdx.x;
  if (idx >= ROWS * NHEADS) return;
  const int row = idx >> 4;
  const int h   = idx & 15;
  const float* qh = q + (size_t)row * DMODEL + h * DK;
  float f[NF];
  float ssum = 0.f;
  #pragma unroll
  for (int n = 0; n < NF; ++n) {
    const float* om = omega + n * DK;
    float dot = 0.f;
    #pragma unroll 8
    for (int k = 0; k < DK; ++k) dot += qh[k] * om[k];
    float e = __expf(-0.5f * dot * dot);
    f[n] = e; ssum += e;
  }
  const float inv = 1.0f / (ssum + 1e-6f);
  float* dst = qp + (size_t)idx * NF;
  #pragma unroll
  for (int n = 0; n < NF; ++n) dst[n] = f[n] * inv;
}

// ---------------------------------------------------------------------------
// Prefix scan over keys: only positions s = 4j+3 are consumed
// (key_idx = (s//4+1)*4-1) -> store only the 1024 sampled states per (b,h).
// ---------------------------------------------------------------------------
__global__ __launch_bounds__(512)
void scan_kernel(const float* __restrict__ kp,    // [ROWS, NHEADS, NF]
                 const float* __restrict__ v,     // [ROWS, DMODEL]
                 float* __restrict__ Ksel,        // [B, SS/4, NHEADS, NF]
                 float* __restrict__ KVsel) {     // [B, SS/4, NHEADS, NF, DK]
  const int b = blockIdx.x >> 4;
  const int h = blockIdx.x & 15;
  const int t = threadIdx.x;
  if (t < NF * DK) {
    const int n = t >> 6, d = t & 63;
    float acc = 0.f;
    for (int s = 0; s < SS; ++s) {
      const size_t rs = (size_t)b * SS + s;
      acc += kp[(rs * NHEADS + h) * NF + n] * v[rs * DMODEL + h * DK + d];
      if ((s & 3) == 3)
        KVsel[((((size_t)b * (SS/4) + (s >> 2)) * NHEADS + h) * NF + n) * DK + d] = acc;
    }
  } else if (t < NF * DK + NF) {
    const int n = t - NF * DK;
    float acc = 0.f;
    for (int s = 0; s < SS; ++s) {
      acc += kp[(((size_t)b * SS + s) * NHEADS + h) * NF + n];
      if ((s & 3) == 3)
        Ksel[(((size_t)b * (SS/4) + (s >> 2)) * NHEADS + h) * NF + n] = acc;
    }
  }
}

// ---------------------------------------------------------------------------
// num/den contraction -> bf16 attention output
// ---------------------------------------------------------------------------
__global__ __launch_bounds__(256)
void numden_kernel(const float* __restrict__ qp,
                   const float* __restrict__ Ksel,
                   const float* __restrict__ KVsel,
                   __bf16* __restrict__ out) {    // [ROWS, DMODEL]
  const size_t idx = (size_t)blockIdx.x * 256 + threadIdx.x;
  if (idx >= (size_t)ROWS * DMODEL) return;
  const int d = idx & 63;
  const size_t t2 = idx >> 6;
  const int h = (int)(t2 & 15);
  const size_t row = t2 >> 4;           // b*SS + s
  const int s = (int)(row & (SS - 1));
  const int b = (int)(row >> 12);       // SS = 4096 = 2^12
  const int j = s >> 2;
  const float* qpr = qp + (row * NHEADS + h) * NF;
  const float* Kr  = Ksel  + (((size_t)b * (SS/4) + j) * NHEADS + h) * NF;
  const float* KVr = KVsel + ((((size_t)b * (SS/4) + j) * NHEADS + h) * NF) * DK + d;
  float num = 0.f, den = 0.f;
  #pragma unroll
  for (int n = 0; n < NF; ++n) {
    const float qv = qpr[n];
    num += qv * KVr[(size_t)n * DK];
    den += qv * Kr[n];
  }
  out[row * DMODEL + h * DK + d] = (__bf16)(num / (den + 1e-6f));
}

// ---------------------------------------------------------------------------
// Host-side orchestration
// ---------------------------------------------------------------------------
extern "C" void kernel_launch(void* const* d_in, const int* in_sizes, int n_in,
                              void* d_out, int out_size, void* d_ws, size_t ws_size,
                              hipStream_t stream) {
  const float* x     = (const float*)d_in[0];
  const float* lto   = (const float*)d_in[1];
  const float* wq_s  = (const float*)d_in[2];
  const float* wk_s  = (const float*)d_in[3];
  const float* wv_s  = (const float*)d_in[4];
  const float* wo_s  = (const float*)d_in[5];
  const float* wq_c  = (const float*)d_in[6];
  const float* wk_c  = (const float*)d_in[7];
  const float* wv_c  = (const float*)d_in[8];
  const float* wo_c  = (const float*)d_in[9];
  const float* omega_s = (const float*)d_in[10];
  const float* omega_c = (const float*)d_in[11];
  const float* w1 = (const float*)d_in[12];
  const float* b1 = (const float*)d_in[13];
  const float* w2 = (const float*)d_in[14];
  const float* b2 = (const float*)d_in[15];
  const float* ln0_a = (const float*)d_in[16];
  const float* ln0_b = (const float*)d_in[17];
  const float* ln1_a = (const float*)d_in[18];
  const float* ln1_b = (const float*)d_in[19];
  const float* ln2_a = (const float*)d_in[20];
  const float* ln2_b = (const float*)d_in[21];
  float* outp = (float*)d_out;

  // bump allocator on workspace (256B aligned)
  char* ws = (char*)d_ws;
  size_t off = 0;
  auto alloc = [&](size_t bytes) -> void* {
    void* p = ws + off;
    off = (off + bytes + 255) & ~(size_t)255;
    return p;
  };

  const size_t MM  = (size_t)DMODEL * DMODEL;    // 1M
  const size_t MF  = (size_t)DFF * DMODEL;       // 4M
  __bf16* wqs16 = (__bf16*)alloc(MM * 2);
  __bf16* wks16 = (__bf16*)alloc(MM * 2);
  __bf16* wvs16 = (__bf16*)alloc(MM * 2);
  __bf16* wos16 = (__bf16*)alloc(MM * 2);
  __bf16* wqc16 = (__bf16*)alloc(MM * 2);
  __bf16* wkc16 = (__bf16*)alloc(MM * 2);
  __bf16* wvc16 = (__bf16*)alloc(MM * 2);
  __bf16* woc16 = (__bf16*)alloc(MM * 2);
  __bf16* w1b16 = (__bf16*)alloc(MF * 2);
  __bf16* w2b16 = (__bf16*)alloc(MF * 2);
  __bf16* xb16   = (__bf16*)alloc((size_t)ROWS * DMODEL * 2);
  __bf16* at16   = (__bf16*)alloc((size_t)ROWS * DMODEL * 2);
  float*  Qf     = (float*)alloc((size_t)ROWS * DMODEL * 4);
  float*  Kf     = (float*)alloc((size_t)ROWS * DMODEL * 4);
  float*  Vf     = (float*)alloc((size_t)ROWS * DMODEL * 4);
  float*  qpb    = (float*)alloc((size_t)ROWS * NHEADS * NF * 4);
  float*  kpb    = (float*)alloc((size_t)ROWS * NHEADS * NF * 4);
  float*  Ksel   = (float*)alloc((size_t)BB * (SS/4) * NHEADS * NF * 4);
  float*  KVsel  = (float*)alloc((size_t)BB * (SS/4) * NHEADS * NF * DK * 4);
  float*  xnew   = (float*)alloc((size_t)ROWS * DMODEL * 4);
  __bf16* xnew16 = (__bf16*)alloc((size_t)ROWS * DMODEL * 2);
  float*  ltonew = (float*)alloc((size_t)ROWS * DMODEL * 4);
  __bf16* h16    = (__bf16*)alloc((size_t)ROWS * DFF * 2);

  auto conv = [&](const float* src, __bf16* dst, size_t n) {
    f2b_kernel<<<(int)((n + 255) / 256), 256, 0, stream>>>(src, dst, (int)n);
  };
  conv(wq_s, wqs16, MM); conv(wk_s, wks16, MM); conv(wv_s, wvs16, MM); conv(wo_s, wos16, MM);
  conv(wq_c, wqc16, MM); conv(wk_c, wkc16, MM); conv(wv_c, wvc16, MM); conv(wo_c, woc16, MM);
  conv(w1, w1b16, MF);   conv(w2, w2b16, MF);

  const dim3 gD(ROWS / 128, DMODEL / 64);   // (64, 16)
  const dim3 gF(ROWS / 128, DFF / 64);      // (64, 64)
  const int phiBlocks = (ROWS * NHEADS + 255) / 256;
  const int ndBlocks  = (int)(((size_t)ROWS * DMODEL) / 256);

  // ---- Stage 1: self performer on x ----
  layernorm_kernel<<<ROWS, 256, 0, stream>>>(x, ln0_a, ln0_b, xb16);
  gemm_bf16_kernel<false,false,false,true,false><<<gD,256,0,stream>>>(xb16, wqs16, nullptr, nullptr, Qf, nullptr, ROWS, DMODEL, DMODEL);
  gemm_bf16_kernel<false,false,false,true,false><<<gD,256,0,stream>>>(xb16, wks16, nullptr, nullptr, Kf, nullptr, ROWS, DMODEL, DMODEL);
  gemm_bf16_kernel<false,false,false,true,false><<<gD,256,0,stream>>>(xb16, wvs16, nullptr, nullptr, Vf, nullptr, ROWS, DMODEL, DMODEL);
  phi_kernel<<<phiBlocks, 256, 0, stream>>>(Qf, omega_s, qpb);
  phi_kernel<<<phiBlocks, 256, 0, stream>>>(Kf, omega_s, kpb);
  scan_kernel<<<BB * NHEADS, 512, 0, stream>>>(kpb, Vf, Ksel, KVsel);
  numden_kernel<<<ndBlocks, 256, 0, stream>>>(qpb, Ksel, KVsel, at16);
  gemm_bf16_kernel<false,false,true,true,true><<<gD,256,0,stream>>>(at16, wos16, nullptr, x, xnew, xnew16, ROWS, DMODEL, DMODEL);

  // ---- Stage 2: cross performer (q from lto, k/v from updated x) ----
  layernorm_kernel<<<ROWS, 256, 0, stream>>>(lto, ln1_a, ln1_b, xb16);
  gemm_bf16_kernel<false,false,false,true,false><<<gD,256,0,stream>>>(xb16,   wqc16, nullptr, nullptr, Qf, nullptr, ROWS, DMODEL, DMODEL);
  gemm_bf16_kernel<false,false,false,true,false><<<gD,256,0,stream>>>(xnew16, wkc16, nullptr, nullptr, Kf, nullptr, ROWS, DMODEL, DMODEL);
  gemm_bf16_kernel<false,false,false,true,false><<<gD,256,0,stream>>>(xnew16, wvc16, nullptr, nullptr, Vf, nullptr, ROWS, DMODEL, DMODEL);
  phi_kernel<<<phiBlocks, 256, 0, stream>>>(Qf, omega_c, qpb);
  phi_kernel<<<phiBlocks, 256, 0, stream>>>(Kf, omega_c, kpb);
  scan_kernel<<<BB * NHEADS, 512, 0, stream>>>(kpb, Vf, Ksel, KVsel);
  numden_kernel<<<ndBlocks, 256, 0, stream>>>(qpb, Ksel, KVsel, at16);
  gemm_bf16_kernel<false,false,true,true,false><<<gD,256,0,stream>>>(at16, woc16, nullptr, lto, ltonew, nullptr, ROWS, DMODEL, DMODEL);

  // ---- Stage 3: FFN ----
  layernorm_kernel<<<ROWS, 256, 0, stream>>>(ltonew, ln2_a, ln2_b, xb16);
  gemm_bf16_kernel<true,true,false,false,true><<<gF,256,0,stream>>>(xb16, w1b16, b1, nullptr, nullptr, h16, ROWS, DFF, DMODEL);
  gemm_bf16_kernel<true,false,true,true,false><<<gD,256,0,stream>>>(h16, w2b16, b2, ltonew, outp, nullptr, ROWS, DMODEL, DFF);

  (void)in_sizes; (void)n_in; (void)out_size; (void)ws_size;
}
